// Raymarcher_27058293965121
// MI455X (gfx1250) — matrix operand order repaired and checked by reading, not compiled, not run on hardware
//
#include <hip/hip_runtime.h>
#include <hip/hip_bf16.h>
#include <math.h>

// ---------------------------------------------------------------------------
// NeRF ray-march + tiny-MLP + composite, fused for gfx1250 (MI455X, wave32).
//   layer1 [3->64]  : v_wmma_f32_16x16x4_f32   (K padded 3->4)
//   layer2 [64->64] : v_wmma_f32_16x16x32_f16  (2 K-steps x 4 N-tiles)
//   heads  [64->4]  : v_wmma_f32_16x16x32_f16  (N padded 4->16, 2 K-steps)
// One workgroup (8 wave32) per ray; all activations live in LDS.
// ---------------------------------------------------------------------------

typedef _Float16 v16h __attribute__((ext_vector_type(16)));
typedef _Float16 v8h  __attribute__((ext_vector_type(8)));
typedef float    v8f  __attribute__((ext_vector_type(8)));
typedef float    v2f  __attribute__((ext_vector_type(2)));

#define NRAYS 8192
#define NSAMP 128
#define HDIM  64
#define GRID3 64
#define LDH   72   // padded LDS row pitch in halves (144B: 16B aligned, conflict-breaking)

__device__ __forceinline__ float sigmoidf_(float x) {
    return 1.0f / (1.0f + __expf(-x));
}

// --- prep kernel: pack f16 weights into workspace
//   ws[0      .. 4095] : W2^T  f16  [n][k]   (B operand for layer2)
//   ws[4096   .. 5119] : Whead^T f16 [n][k]  n=0:Wsig, n=1..3:Wrgb col, n>=4: 0
__global__ void prep_weights_f16(const float* __restrict__ W2,
                                 const float* __restrict__ Wsig,
                                 const float* __restrict__ Wrgb,
                                 _Float16* __restrict__ ws) {
    int id = blockIdx.x * blockDim.x + threadIdx.x;
    if (id < HDIM * HDIM) {
        int n = id >> 6;
        int k = id & 63;
        ws[id] = (_Float16)W2[k * HDIM + n];
    } else if (id < HDIM * HDIM + 16 * HDIM) {
        int hid = id - HDIM * HDIM;
        int n = hid >> 6;
        int k = hid & 63;
        float v = 0.0f;
        if (n == 0)      v = Wsig[k];
        else if (n < 4)  v = Wrgb[k * 3 + (n - 1)];
        ws[id] = (_Float16)v;
    }
}

__global__ __launch_bounds__(256)
void nerf_fused_kernel(const float* __restrict__ rays_o,
                       const float* __restrict__ rays_d,
                       const float* __restrict__ nearv,
                       const float* __restrict__ farv,
                       const float* __restrict__ jitter,
                       const float* __restrict__ density,
                       const float* __restrict__ W1,
                       const float* __restrict__ b1,
                       const float* __restrict__ b2,
                       const float* __restrict__ bsig,
                       const float* __restrict__ brgb,
                       const _Float16* __restrict__ wpk,
                       float* __restrict__ out) {
    __shared__ _Float16 sW2T[HDIM * LDH];    //  9216 B  [n][k]
    __shared__ _Float16 sWHT[16 * LDH];      //  2304 B  [n][k] heads
    __shared__ _Float16 sH1[NSAMP * LDH];    // 18432 B  [s][k]
    __shared__ _Float16 sH2[NSAMP * LDH];    // 18432 B  [s][k]
    __shared__ float sW1[3 * HDIM];
    __shared__ float sB1[HDIM];
    __shared__ float sB2[HDIM];
    __shared__ float sPts[NSAMP * 3];
    __shared__ float sAlpha[NSAMP];
    __shared__ float sRgb[NSAMP * 3];
    __shared__ int   sMask[NSAMP];

    const int r   = blockIdx.x;       // one ray per workgroup
    const int tid = threadIdx.x;      // 256 threads = 8 wave32

    __builtin_prefetch(density, 0, 1);

    // ---------------- Phase A: stage weights into LDS ----------------
    {
        int n   = tid >> 2;           // 0..63
        int seg = tid & 3;
        const v8h* src = (const v8h*)(wpk + n * HDIM + seg * 16);
        v8h* dst = (v8h*)(&sW2T[n * LDH + seg * 16]);
        dst[0] = src[0];
        dst[1] = src[1];
    }
    if (tid < 64) {                   // heads: 16 rows x 64 halves
        int n   = tid >> 2;           // 0..15
        int seg = tid & 3;
        const v8h* src = (const v8h*)(wpk + HDIM * HDIM + n * HDIM + seg * 16);
        v8h* dst = (v8h*)(&sWHT[n * LDH + seg * 16]);
        dst[0] = src[0];
        dst[1] = src[1];
    }
    if (tid < 3 * HDIM) sW1[tid] = W1[tid];
    if (tid < HDIM)     { sB1[tid] = b1[tid]; sB2[tid] = b2[tid]; }

    // ---------------- Phase B: ray march / occupancy / jitter ----------------
    const float o0 = rays_o[r * 3 + 0], o1 = rays_o[r * 3 + 1], o2 = rays_o[r * 3 + 2];
    const float d0 = rays_d[r * 3 + 0], d1 = rays_d[r * 3 + 1], d2 = rays_d[r * 3 + 2];
    const float zn = nearv[r];
    const float step = (farv[r] - zn) * (1.0f / (float)NSAMP);

    if (tid < NSAMP) {
        const int s = tid;
        float z = zn + (float)s * step;
        float p0 = o0 + z * d0, p1 = o1 + z * d1, p2 = o2 + z * d2;
        // AABB_MIN = (-1.25,-1.55,-1.25), scale = 2.5 all dims
        float u0 = (p0 + 1.25f) * (1.0f / 2.5f);
        float u1 = (p1 + 1.55f) * (1.0f / 2.5f);
        float u2 = (p2 + 1.25f) * (1.0f / 2.5f);
        int i0 = (int)floorf(u0 * (float)GRID3);
        int i1 = (int)floorf(u1 * (float)GRID3);
        int i2 = (int)floorf(u2 * (float)GRID3);
        bool inb = (i0 >= 0) && (i0 < GRID3) && (i1 >= 0) && (i1 < GRID3) &&
                   (i2 >= 0) && (i2 < GRID3);
        int c0 = min(max(i0, 0), GRID3 - 1);
        int c1 = min(max(i1, 0), GRID3 - 1);
        int c2 = min(max(i2, 0), GRID3 - 1);
        float dv = density[(c0 * GRID3 + c1) * GRID3 + c2];
        bool occ = dv > 0.5f;
        float zval = (occ && inb) ? z : 0.0f;
        int msk = zval > 0.0f;
        float zj = zval + jitter[r * NSAMP + s] * step;
        sPts[s * 3 + 0] = o0 + zj * d0;
        sPts[s * 3 + 1] = o1 + zj * d1;
        sPts[s * 3 + 2] = o2 + zj * d2;
        sMask[s] = msk;
    }
    __syncthreads();

    const int lane = tid & 31;
    const int lm   = lane & 15;
    const bool hi  = lane >= 16;
    const int m0   = (tid >> 5) * 16;         // this wave's 16 sample rows
    const int Mrow = m0 + lm;

    // ---------------- Phase C: layer 1 on WMMA f32 16x16x4 (K=3 pad 4) -------
    {
        // A frag (16x4 f32): lanes 0-15 hold K=0,1; lanes 16-31 hold K=2,3(pad)
        v2f a;
        a[0] = hi ? sPts[Mrow * 3 + 2] : sPts[Mrow * 3 + 0];
        a[1] = hi ? 0.0f               : sPts[Mrow * 3 + 1];
        #pragma unroll
        for (int nt = 0; nt < 4; ++nt) {
            const int N = nt * 16 + lm;
            v2f b;
            b[0] = hi ? sW1[2 * HDIM + N] : sW1[0 * HDIM + N];
            b[1] = hi ? 0.0f              : sW1[1 * HDIM + N];
            v8f acc = {};
            acc = __builtin_amdgcn_wmma_f32_16x16x4_f32(
                    false, a, false, b, (short)0, acc, false, false);
            #pragma unroll
            for (int e = 0; e < 8; ++e) {
                int row = m0 + e + (hi ? 8 : 0);
                float v = fmaxf(acc[e] + sB1[N], 0.0f);
                sH1[row * LDH + N] = (_Float16)v;
            }
        }
    }
    __syncthreads();

    // ---------------- Phase D: layer 2 on WMMA f16 (128x64 @ 64x64) ----------
    {
        const _Float16* h1row = &sH1[Mrow * LDH];
        #pragma unroll
        for (int nt = 0; nt < 4; ++nt) {
            const int Ncol = nt * 16 + lm;
            const _Float16* brow = &sW2T[Ncol * LDH];
            v8f acc = {};
            #pragma unroll
            for (int kk = 0; kk < 64; kk += 32) {
                // A frag: lanes 0-15 hold K {0..7,16..23}, lanes 16-31 {8..15,24..31}
                int kb = kk + (hi ? 8 : 0);
                v8h a0 = *(const v8h*)(h1row + kb);
                v8h a1 = *(const v8h*)(h1row + kb + 16);
                v16h a;
                #pragma unroll
                for (int e = 0; e < 8; ++e) { a[e] = a0[e]; a[e + 8] = a1[e]; }
                // B frag: lanes 0-15 hold K {0..7,16..23}, lanes 16-31 {8..15,24..31}
                int ko = kk + (hi ? 8 : 0);
                v8h bb0 = *(const v8h*)(brow + ko);
                v8h bb1 = *(const v8h*)(brow + ko + 16);
                v16h b;
                #pragma unroll
                for (int e = 0; e < 8; ++e) { b[e] = bb0[e]; b[e + 8] = bb1[e]; }
                acc = __builtin_amdgcn_wmma_f32_16x16x32_f16(
                        false, a, false, b, (short)0, acc, false, false);
            }
            #pragma unroll
            for (int e = 0; e < 8; ++e) {
                int row = m0 + e + (hi ? 8 : 0);
                float v = fmaxf(acc[e] + sB2[Ncol], 0.0f);
                sH2[row * LDH + Ncol] = (_Float16)v;
            }
        }
    }
    __syncthreads();

    // ---------------- Phase E: heads on WMMA f16 (N padded 4->16) ------------
    {
        const _Float16* h2row = &sH2[Mrow * LDH];
        const _Float16* brow  = &sWHT[lm * LDH];
        v8f acc = {};
        #pragma unroll
        for (int kk = 0; kk < 64; kk += 32) {
            int kb = kk + (hi ? 8 : 0);
            v8h a0 = *(const v8h*)(h2row + kb);
            v8h a1 = *(const v8h*)(h2row + kb + 16);
            v16h a;
            #pragma unroll
            for (int e = 0; e < 8; ++e) { a[e] = a0[e]; a[e + 8] = a1[e]; }
            int ko = kk + (hi ? 8 : 0);
            v8h bb0 = *(const v8h*)(brow + ko);
            v8h bb1 = *(const v8h*)(brow + ko + 16);
            v16h b;
            #pragma unroll
            for (int e = 0; e < 8; ++e) { b[e] = bb0[e]; b[e + 8] = bb1[e]; }
            acc = __builtin_amdgcn_wmma_f32_16x16x32_f16(
                    false, a, false, b, (short)0, acc, false, false);
        }
        // col 0 -> sigma -> alpha; cols 1..3 -> rgb (divergence AFTER the wmma)
        if (lm < 4) {
            float bias = (lm == 0) ? bsig[0] : brgb[lm - 1];
            #pragma unroll
            for (int e = 0; e < 8; ++e) {
                int row = m0 + e + (hi ? 8 : 0);
                int msk = sMask[row];
                float v = acc[e] + bias;
                if (lm == 0) {
                    // masked -> sigma=-1000 -> relu 0 -> alpha 0
                    float tau = msk ? fmaxf(v, 0.0f) * step : 0.0f;
                    sAlpha[row] = 1.0f - __expf(-tau);
                } else {
                    sRgb[row * 3 + (lm - 1)] = msk ? sigmoidf_(v) : 0.0f;
                }
            }
        }
    }
    __syncthreads();

    // ---------------- Phase F: ordered transmittance composite ---------------
    if (tid == 0) {
        float T = 1.0f, cr = 0.0f, cg = 0.0f, cb = 0.0f;
        for (int s = 0; s < NSAMP; ++s) {
            float alpha = sAlpha[s];
            float w     = alpha * T;
            cr += w * sRgb[s * 3 + 0];
            cg += w * sRgb[s * 3 + 1];
            cb += w * sRgb[s * 3 + 2];
            T *= (1.0f - alpha + 1e-10f);
        }
        out[r * 3 + 0] = cr + T;   // white background: + trans[-1]
        out[r * 3 + 1] = cg + T;
        out[r * 3 + 2] = cb + T;
    }
}

extern "C" void kernel_launch(void* const* d_in, const int* in_sizes, int n_in,
                              void* d_out, int out_size, void* d_ws, size_t ws_size,
                              hipStream_t stream) {
    const float* rays_o  = (const float*)d_in[0];
    const float* rays_d  = (const float*)d_in[1];
    const float* nearv   = (const float*)d_in[2];
    const float* farv    = (const float*)d_in[3];
    const float* jitter  = (const float*)d_in[4];
    const float* density = (const float*)d_in[5];
    const float* W1      = (const float*)d_in[6];
    const float* b1      = (const float*)d_in[7];
    const float* W2      = (const float*)d_in[8];
    const float* b2      = (const float*)d_in[9];
    const float* Wsig    = (const float*)d_in[10];
    const float* bsig    = (const float*)d_in[11];
    const float* Wrgb    = (const float*)d_in[12];
    const float* brgb    = (const float*)d_in[13];
    float* out = (float*)d_out;

    _Float16* wpk = (_Float16*)d_ws;   // 5120 halves = 10 KB

    prep_weights_f16<<<dim3(20), dim3(256), 0, stream>>>(W2, Wsig, Wrgb, wpk);
    nerf_fused_kernel<<<dim3(NRAYS), dim3(256), 0, stream>>>(
        rays_o, rays_d, nearv, farv, jitter, density,
        W1, b1, b2, bsig, brgb, wpk, out);
}
